// SelfAttention_4827543241009
// MI455X (gfx1250) — matrix operand order repaired and checked by reading, not compiled, not run on hardware
//
#include <hip/hip_runtime.h>
#include <math.h>

// ---------------------------------------------------------------------------
// Fused causal self-attention for MI455X (gfx1250, wave32, WMMA bf16)
// B=1024, T=256, C=384, H=64.  One block per batch, 16 waves, 16 q-rows/wave.
// ---------------------------------------------------------------------------

typedef __attribute__((ext_vector_type(16))) __bf16 bf16x16;
typedef __attribute__((ext_vector_type(8)))  float  f32x8;

#define WMMA_BF16(a, b, c) \
    __builtin_amdgcn_wmma_f32_16x16x32_bf16(false, (a), false, (b), (short)0, (c), false, false)

static constexpr int Bn = 1024;
static constexpr int Tn = 256;
static constexpr int Cn = 384;
static constexpr int Hn = 64;
static constexpr float SCALE = 0.05103103630798287f; // 384^-0.5
static constexpr float NEGBIG = -1.0e30f;

// ----- A fragment (16x32 bf16) from row-major fp32 global matrix ------------
// Lane l: M = l&15. half = l>>4.
//   VGPR j in 0..3: K = half*16 + 2j, 2j+1
//   VGPR j in 4..7: K = half*16 + 8 + 2(j-4), +1
__device__ __forceinline__ bf16x16 load_a_f32(const float* base, int ld,
                                              int row0, int k0, int lane) {
    const int m = lane & 15, half = lane >> 4;
    const float* r = base + (size_t)(row0 + m) * ld + k0;
    bf16x16 a;
#pragma unroll
    for (int j = 0; j < 4; ++j) {
        int k = half * 16 + 2 * j;
        a[2 * j]     = (__bf16)r[k];
        a[2 * j + 1] = (__bf16)r[k + 1];
    }
#pragma unroll
    for (int j = 0; j < 4; ++j) {
        int k = half * 16 + 8 + 2 * j;
        a[8 + 2 * j]     = (__bf16)r[k];
        a[8 + 2 * j + 1] = (__bf16)r[k + 1];
    }
    return a;
}

// ----- A fragment (16x32 bf16) from row-major bf16 LDS matrix ---------------
__device__ __forceinline__ bf16x16 load_a_lds(const __bf16* base, int ld,
                                              int row0, int k0, int lane) {
    const int m = lane & 15, half = lane >> 4;
    const __bf16* r = base + (row0 + m) * ld + k0;
    bf16x16 a;
#pragma unroll
    for (int j = 0; j < 4; ++j) {
        int k = half * 16 + 2 * j;
        a[2 * j]     = r[k];
        a[2 * j + 1] = r[k + 1];
    }
#pragma unroll
    for (int j = 0; j < 4; ++j) {
        int k = half * 16 + 8 + 2 * j;
        a[8 + 2 * j]     = r[k];
        a[8 + 2 * j + 1] = r[k + 1];
    }
    return a;
}

// ----- B fragment (32x16 bf16) from K-major fp32 global matrix W[K][N] ------
// Lanes 0..15 hold K=0..15 (VGPR j packs K=2j,2j+1), lanes 16..31 hold K=16..31.
__device__ __forceinline__ bf16x16 load_b_f32(const float* base, int ld,
                                              int k0, int n0, int lane) {
    const int n = lane & 15;
    const int kb = k0 + ((lane >> 4) << 4);
    bf16x16 b;
#pragma unroll
    for (int j = 0; j < 8; ++j) {
        b[2 * j]     = (__bf16)base[(size_t)(kb + 2 * j)     * ld + n0 + n];
        b[2 * j + 1] = (__bf16)base[(size_t)(kb + 2 * j + 1) * ld + n0 + n];
    }
    return b;
}

// ----- B fragment (32x16 bf16) from LDS stored N-major: value(k,n)=buf[n*ld+k]
// (used for K^T from k_smem[key][h] and for V from vT_smem[h][key])
__device__ __forceinline__ bf16x16 load_b_lds_t(const __bf16* buf, int ld,
                                                int k0, int n0, int lane) {
    const int n = lane & 15;
    const int kb = k0 + ((lane >> 4) << 4);
    const __bf16* r = buf + (n0 + n) * ld + kb;
    bf16x16 b;
#pragma unroll
    for (int j = 0; j < 16; ++j) b[j] = r[j];
    return b;
}

extern __shared__ __bf16 smem[];

__global__ __launch_bounds__(512)
void attn_fused_kernel(const float* __restrict__ idx,
                       const float* __restrict__ Wq,
                       const float* __restrict__ Wk,
                       const float* __restrict__ Wv,
                       float* __restrict__ out) {
    // LDS carve-up (bf16 elements)
    __bf16* q_smem  = smem;                       // [256][64]  32KB
    __bf16* k_smem  = q_smem + Tn * Hn;           // [256][64]  32KB
    __bf16* vT_smem = k_smem + Tn * Hn;           // [64][256]  32KB  V^T
    __bf16* p_smem  = vT_smem + Hn * Tn;          // 16 waves x [16][32] 16KB

    const int wave = threadIdx.x >> 5;
    const int lane = threadIdx.x & 31;
    const int nIdx = lane & 15;
    const int mhi  = (lane >> 4) << 3;

    const float* x = idx + (size_t)blockIdx.x * Tn * Cn;

    // ---------------- Phase 1: QKV projection into LDS (WMMA bf16) ----------
    // 192 output tiles: p=0 -> Q, p=1 -> K, p=2 -> V (transposed store)
    for (int t = wave; t < 192; t += 16) {
        const int p    = t >> 6;
        const int r    = t & 63;
        const int row0 = (r >> 2) << 4;   // 0..240
        const int n0   = (r & 3) << 4;    // 0..48
        const float* W = (p == 0) ? Wq : (p == 1) ? Wk : Wv;

        f32x8 acc = {};
#pragma unroll
        for (int k0 = 0; k0 < Cn; k0 += 32) {
            bf16x16 a = load_a_f32(x, Cn, row0, k0, lane);
            bf16x16 b = load_b_f32(W, Hn, k0, n0, lane);
            acc = WMMA_BF16(a, b, acc);
        }
#pragma unroll
        for (int v = 0; v < 8; ++v) {
            const int m = row0 + v + mhi;
            const __bf16 val = (__bf16)acc[v];
            if (p == 0)      q_smem[m * Hn + n0 + nIdx] = val;
            else if (p == 1) k_smem[m * Hn + n0 + nIdx] = val;
            else             vT_smem[(n0 + nIdx) * Tn + m] = val;
        }
    }
    __syncthreads();

    // ---------------- Phase 2: flash attention, 16 q-rows per wave ----------
    const int q0 = wave * 16;

    float m_run[8], l_run[8];
#pragma unroll
    for (int v = 0; v < 8; ++v) { m_run[v] = NEGBIG; l_run[v] = 0.f; }
    f32x8 o[4] = {{}, {}, {}, {}};

    // Q fragments for this wave's 16 rows (K = H = 64 -> two 16x32 frags)
    const bf16x16 aq0 = load_a_lds(q_smem, Hn, q0, 0, lane);
    const bf16x16 aq1 = load_a_lds(q_smem, Hn, q0, 32, lane);

    __bf16* pw = p_smem + wave * (16 * 32);

    const int nchunks = (q0 + 15) / 32 + 1;   // causal early exit
    for (int c = 0; c < nchunks; ++c) {
        // ---- scores: two 16x16 tiles over keys [32c, 32c+32) ----
        f32x8 s[2];
#pragma unroll
        for (int t = 0; t < 2; ++t) {
            const int kcol = c * 32 + t * 16;
            f32x8 acc = {};
            acc  = WMMA_BF16(aq0, load_b_lds_t(k_smem, Hn, 0,  kcol, lane), acc);
            acc  = WMMA_BF16(aq1, load_b_lds_t(k_smem, Hn, 32, kcol, lane), acc);
            s[t] = acc;
        }

        // ---- scale + causal mask + chunk row-max ----
        float cmax[8];
#pragma unroll
        for (int v = 0; v < 8; ++v) {
            const int mrow = q0 + v + mhi;
#pragma unroll
            for (int t = 0; t < 2; ++t) {
                const int key = c * 32 + t * 16 + nIdx;
                float xv = s[t][v] * SCALE;
                s[t][v] = (key <= mrow) ? xv : NEGBIG;
            }
            cmax[v] = fmaxf(s[0][v], s[1][v]);
        }
#pragma unroll
        for (int v = 0; v < 8; ++v) {
#pragma unroll
            for (int d = 1; d < 16; d <<= 1)
                cmax[v] = fmaxf(cmax[v], __shfl_xor(cmax[v], d, 32));
        }

        // ---- online softmax update ----
        float alpha[8], csum[8];
#pragma unroll
        for (int v = 0; v < 8; ++v) {
            const float nm = fmaxf(m_run[v], cmax[v]);
            alpha[v] = __expf(m_run[v] - nm);
            m_run[v] = nm;
            const float p0 = __expf(s[0][v] - nm);
            const float p1 = __expf(s[1][v] - nm);
            s[0][v] = p0; s[1][v] = p1;
            csum[v] = p0 + p1;
        }
#pragma unroll
        for (int v = 0; v < 8; ++v) {
#pragma unroll
            for (int d = 1; d < 16; d <<= 1)
                csum[v] += __shfl_xor(csum[v], d, 32);
            l_run[v] = l_run[v] * alpha[v] + csum[v];
        }
#pragma unroll
        for (int nh = 0; nh < 4; ++nh)
#pragma unroll
            for (int v = 0; v < 8; ++v) o[nh][v] *= alpha[v];

        // ---- transpose P (C-layout) -> A-layout via per-wave LDS scratch ----
#pragma unroll
        for (int v = 0; v < 8; ++v) {
            pw[(v + mhi) * 32 + nIdx]      = (__bf16)s[0][v];
            pw[(v + mhi) * 32 + 16 + nIdx] = (__bf16)s[1][v];
        }
        asm volatile("s_wait_dscnt 0x0" ::: "memory"); // per-wave DS RAW fence

        const bf16x16 ap = load_a_lds(pw, 32, 0, 0, lane);
#pragma unroll
        for (int nh = 0; nh < 4; ++nh)
            o[nh] = WMMA_BF16(ap, load_b_lds_t(vT_smem, Tn, c * 32, nh * 16, lane), o[nh]);
    }

    // ---------------- epilogue: normalize + write fp32 ----------------------
    float* op = out + (size_t)blockIdx.x * Tn * Hn;
#pragma unroll
    for (int nh = 0; nh < 4; ++nh)
#pragma unroll
        for (int v = 0; v < 8; ++v)
            op[(size_t)(q0 + v + mhi) * Hn + nh * 16 + nIdx] = o[nh][v] / l_run[v];
}

extern "C" void kernel_launch(void* const* d_in, const int* in_sizes, int n_in,
                              void* d_out, int out_size, void* d_ws, size_t ws_size,
                              hipStream_t stream) {
    const float* idx = (const float*)d_in[0];
    const float* Wq  = (const float*)d_in[1];
    const float* Wk  = (const float*)d_in[2];
    const float* Wv  = (const float*)d_in[3];
    float* out = (float*)d_out;

    const size_t smem_bytes = (size_t)(Tn * Hn + Tn * Hn + Hn * Tn + 16 * 16 * 32) * sizeof(__bf16); // 112KB
    static_assert((Tn * Hn + Tn * Hn + Hn * Tn + 16 * 16 * 32) * sizeof(__bf16) == 114688, "lds size");

    // Raise dynamic-LDS cap above the 64KB default (gfx1250 WGP has 320KB).
    (void)hipFuncSetAttribute((const void*)attn_fused_kernel,
                              hipFuncAttributeMaxDynamicSharedMemorySize,
                              (int)smem_bytes);

    attn_fused_kernel<<<dim3(Bn), dim3(512), smem_bytes, stream>>>(idx, Wq, Wk, Wv, out);
}
